// MultiHeadAttention_48395691492077
// MI455X (gfx1250) — compile-verified, matrix operand order checked by reading, not run on hardware
//
#include <hip/hip_runtime.h>
#include <hip/hip_bf16.h>

// ---------------------------------------------------------------------------
// Transformer block for B=4, S=2048, D=1024 on gfx1250 (MI455X).
// - All GEMMs: V_WMMA_F32_16X16X32_BF16 (wave32, f32 accumulate).
// - GEMM staging: TDM tensor_load_to_lds (TENSORcnt) if available, else
//   async global->LDS (ASYNCcnt), else sync load+ds_store. Double-buffered.
// ---------------------------------------------------------------------------

typedef __attribute__((ext_vector_type(16))) __bf16 v16bf;
typedef __attribute__((ext_vector_type(8)))  __bf16 v8bf;
typedef __attribute__((ext_vector_type(4)))  __bf16 v4bf;
typedef __attribute__((ext_vector_type(8)))  float  v8f;
typedef int i32x4 __attribute__((ext_vector_type(4)));

#define ASG __attribute__((address_space(1)))
#define ASL __attribute__((address_space(3)))

#if defined(__has_builtin)
#  if __has_builtin(__builtin_amdgcn_tensor_load_to_lds)
#    define HAVE_TDM 1
#  endif
#  if __has_builtin(__builtin_amdgcn_global_load_async_to_lds_b128)
#    define HAVE_ASYNC_LDS 1
#  endif
#endif

#if __has_include(<hip/amd_detail/amd_gfx1250_TDM.h>)
#  define TDM_ARGS6 1   // therock-10.0 headers -> 6-arg builtin
#endif

constexpr int Bc   = 4;
constexpr int Sc   = 2048;
constexpr int Dc   = 1024;
constexpr int Mtok = Bc * Sc;     // 8192 token rows
constexpr int NQKV = 3 * Dc;      // 3072

// ---- WMMA fragment loader -------------------------------------------------
// bf16 A (16x32) / B-as-[N,K] fragment: per lane, two contiguous 8-bf16 (16B)
// chunks at k-offsets {half*8, 16+half*8}  (ISA 7.12.2 16-bit A layout).
__device__ __forceinline__ v16bf load_frag(const __bf16* __restrict__ row, int half) {
  v8bf lo = *reinterpret_cast<const v8bf*>(row + half * 8);
  v8bf hi = *reinterpret_cast<const v8bf*>(row + 16 + half * 8);
  v16bf r;
#pragma unroll
  for (int i = 0; i < 8; ++i) { r[i] = lo[i]; r[i + 8] = hi[i]; }
  return r;
}

__device__ __forceinline__ v8f wmma_bf16(v16bf a, v16bf b, v8f c) {
  return __builtin_amdgcn_wmma_f32_16x16x32_bf16(
      /*neg_a=*/false, a, /*neg_b=*/false, b,
      /*c_mod=*/(short)0, c, /*reuse_a=*/false, /*reuse_b=*/false);
}

// ---- LDS byte offset of a __shared__ object -------------------------------
__device__ __forceinline__ unsigned lds_off(const void* p) {
  return (unsigned)(unsigned long long)(ASL const char*)p;
}

// ---- TDM 2D tile load descriptor (ISA 8.3/8.4) ----------------------------
#if defined(HAVE_TDM)
typedef unsigned int tdmu4 __attribute__((ext_vector_type(4)));
typedef int          tdmi8 __attribute__((ext_vector_type(8)));
typedef int          tdmi4 __attribute__((ext_vector_type(4)));

// Loads tile [tile1 rows x tile0 elems] of bf16 from row-major global (row
// stride = stride0 elems) into LDS at lds_byte, padding LDS rows 64B->80B
// (pad_interval=3: 16 dwords, pad_amount=3: 4 dwords) to match LSH=40.
__device__ __forceinline__ void tdm_load_2d(const __bf16* g, unsigned lds_byte,
                                            unsigned tile0, unsigned tile1,
                                            unsigned stride0_elems) {
  const unsigned long long ga = (unsigned long long)g;
  tdmu4 g0;
  g0[0] = 1u;                                                  // count=1, user
  g0[1] = lds_byte;                                            // lds_addr
  g0[2] = (unsigned)ga;                                        // gaddr[31:0]
  g0[3] = (unsigned)((ga >> 32) & 0x01FFFFFFull) | (2u << 30); // gaddr[56:32]|type=2
  tdmi8 g1;
  g1[0] = (int)((1u << 16)       // data_size = 1 -> 2 bytes (bf16)
              | (1u << 20)       // pad_enable
              | (3u << 22)       // pad_interval: 16 dwords (64B)
              | (3u << 25));     // pad_amount:   4 dwords (16B)
  g1[1] = (int)((tile0 & 0xFFFFu) << 16);                         // tensor_dim0 lo
  g1[2] = (int)(((tile0 >> 16) & 0xFFFFu) | ((tile1 & 0xFFFFu) << 16)); // dim0 hi|dim1 lo
  g1[3] = (int)(((tile1 >> 16) & 0xFFFFu) | ((tile0 & 0xFFFFu) << 16)); // dim1 hi|tile_dim0
  g1[4] = (int)(tile1 & 0xFFFFu);                                 // tile_dim1; tile_dim2=0
  g1[5] = (int)stride0_elems;                                     // dim0_stride[31:0]
  g1[6] = 0;                                                      // dim0_stride[47:32]
  g1[7] = 0;
  tdmi4 z4 = {0, 0, 0, 0};
#  if defined(TDM_ARGS6)
  tdmi8 z8 = {0, 0, 0, 0, 0, 0, 0, 0};
  __builtin_amdgcn_tensor_load_to_lds(g0, g1, z4, z4, z8, 0);
#  else
  __builtin_amdgcn_tensor_load_to_lds(g0, g1, z4, z4, 0);
#  endif
}
#endif  // HAVE_TDM

// ---- per-thread 64B row staging (async if available) ----------------------
__device__ __forceinline__ void stage_row(const __bf16* g, __bf16* l) {
#if defined(HAVE_ASYNC_LDS)
#pragma unroll
  for (int q = 0; q < 4; ++q)
    __builtin_amdgcn_global_load_async_to_lds_b128(
        (i32x4*)(g + q * 8), (i32x4*)(l + q * 8), 0, 0);
#else
  __builtin_prefetch(g + 32, 0, 0);
  uint4 r[4];
#pragma unroll
  for (int q = 0; q < 4; ++q) r[q] = reinterpret_cast<const uint4*>(g)[q];
#pragma unroll
  for (int q = 0; q < 4; ++q) reinterpret_cast<uint4*>(l)[q] = r[q];
#endif
}

__device__ __forceinline__ void stage_wait(int wave) {
#if defined(HAVE_TDM)
  if (wave == 0) __builtin_amdgcn_s_wait_tensorcnt(0);
#elif defined(HAVE_ASYNC_LDS)
#  if __has_builtin(__builtin_amdgcn_s_wait_asynccnt)
  __builtin_amdgcn_s_wait_asynccnt(0);
#  else
  asm volatile("s_wait_asynccnt 0x0" ::: "memory");
#  endif
#else
  (void)wave;
#endif
}

// ---- fp32 -> bf16 convert -------------------------------------------------
__global__ __launch_bounds__(256) void cvt_bf16_kernel(
    const float* __restrict__ src, __bf16* __restrict__ dst, long n) {
  long i = ((long)blockIdx.x * blockDim.x + threadIdx.x) * 4;
  if (i + 3 < n) {
    float4 f = *reinterpret_cast<const float4*>(src + i);
    v4bf o;
    o[0] = (__bf16)f.x; o[1] = (__bf16)f.y; o[2] = (__bf16)f.z; o[3] = (__bf16)f.w;
    *reinterpret_cast<v4bf*>(dst + i) = o;
  }
}

// ---- fp32 [K,N] -> bf16 [N,K] transpose-convert ---------------------------
__global__ __launch_bounds__(256) void transpose_cvt_kernel(
    const float* __restrict__ src, __bf16* __restrict__ dst, int K, int N) {
  __shared__ float tile[32][33];
  const int nb = blockIdx.x * 32;
  const int kb = blockIdx.y * 32;
  const int tx = threadIdx.x & 31;
  const int ty = threadIdx.x >> 5;   // 0..7
#pragma unroll
  for (int i = ty; i < 32; i += 8)
    tile[i][tx] = src[(size_t)(kb + i) * N + (nb + tx)];
  __syncthreads();
#pragma unroll
  for (int i = ty; i < 32; i += 8)
    dst[(size_t)(nb + i) * K + (kb + tx)] = (__bf16)tile[tx][i];
}

// ---- tiled bf16 GEMM, double-buffered DMA pipeline ------------------------
// C[M,N] = A[M,K](bf16,row-major) @ Bt[N,K]^T  (+bias)(+residual)(relu?)
// block: 128 threads = 4 waves; tile 128x128; wave tile 64x64 (4x4 WMMA).
#define TM  128
#define TN  128
#define TKg 32
#define LSH 40   // padded LDS row stride (bf16): 80B, 16B-aligned, conflict-free

__global__ __launch_bounds__(128) void gemm_bf16_kernel(
    const __bf16* __restrict__ A,  int lda,
    const __bf16* __restrict__ Bt, int ldb,
    int K,
    const float* __restrict__ bias,
    const float* __restrict__ resid,
    int do_relu,
    float*  __restrict__ outF,
    __bf16* __restrict__ outB,
    int ldc) {
  __shared__ __bf16 sA[2][TM * LSH];
  __shared__ __bf16 sB[2][TN * LSH];

  const int m0   = blockIdx.x * TM;
  const int n0   = blockIdx.y * TN;
  const int tid  = threadIdx.x;
  const int wave = tid >> 5;
  const int lane = tid & 31;
  const int lm   = lane & 15;
  const int lh   = lane >> 4;
  const int wm   = (wave >> 1) * 64;
  const int wn   = (wave & 1) * 64;

  v8f zero8 = {0.f, 0.f, 0.f, 0.f, 0.f, 0.f, 0.f, 0.f};
  v8f acc[4][4];
#pragma unroll
  for (int i = 0; i < 4; ++i)
#pragma unroll
    for (int j = 0; j < 4; ++j) acc[i][j] = zero8;

  auto stage = [&](int buf, int k0) {
#if defined(HAVE_TDM)
    if (wave == 0) {  // one DMA issue per tile; EXEC-independent engine op
      tdm_load_2d(A  + (size_t)m0 * lda + k0, lds_off(&sA[buf][0]), TKg, TM, (unsigned)lda);
      tdm_load_2d(Bt + (size_t)n0 * ldb + k0, lds_off(&sB[buf][0]), TKg, TN, (unsigned)ldb);
    }
#else
    stage_row(A  + (size_t)(m0 + tid) * lda + k0, &sA[buf][tid * LSH]);
    stage_row(Bt + (size_t)(n0 + tid) * ldb + k0, &sB[buf][tid * LSH]);
#endif
  };

  stage(0, 0);
  int buf = 0;
  for (int k0 = 0; k0 < K; k0 += TKg) {
    stage_wait(wave);
    __syncthreads();                     // publish DMA'd tile to all waves
    if (k0 + TKg < K) stage(buf ^ 1, k0 + TKg);  // overlap next tile's copy

    v16bf af[4], bfr[4];
#pragma unroll
    for (int i = 0; i < 4; ++i)
      af[i] = load_frag(&sA[buf][(wm + i * 16 + lm) * LSH], lh);
#pragma unroll
    for (int j = 0; j < 4; ++j)
      bfr[j] = load_frag(&sB[buf][(wn + j * 16 + lm) * LSH], lh);

#pragma unroll
    for (int i = 0; i < 4; ++i)
#pragma unroll
      for (int j = 0; j < 4; ++j)
        acc[i][j] = wmma_bf16(af[i], bfr[j], acc[i][j]);
    buf ^= 1;
  }

  // epilogue: D layout -> row = (i*16 + lh*8 + r), col = (j*16 + lm)
#pragma unroll
  for (int i = 0; i < 4; ++i) {
#pragma unroll
    for (int j = 0; j < 4; ++j) {
      const int col = n0 + wn + j * 16 + lm;
      const float bv = bias ? bias[col] : 0.0f;
#pragma unroll
      for (int r = 0; r < 8; ++r) {
        const size_t rr = (size_t)(m0 + wm + i * 16 + lh * 8 + r);
        float v = acc[i][j][r] + bv;
        if (resid)   v += resid[rr * ldc + col];
        if (do_relu) v = v > 0.f ? v : 0.f;
        if (outF) outF[rr * ldc + col] = v;
        if (outB) outB[rr * ldc + col] = (__bf16)v;
      }
    }
  }
}

// ---- flash attention ------------------------------------------------------
// Block = 16 queries of one batch; 4 waves each own a 256-wide D slice.
// Mask semantics (faithful to reference): s = qk/2; keep t>=s (triu);
// any exact zero -> -9e15; softmax over t; out = P @ V.
#define DSLICE 256
#define SMEM_ATTN (8192 + 192 + 1280 + 4 * DSLICE * 32 * 2)  // 75200 B

__global__ __launch_bounds__(128) void attn_flash_kernel(
    const __bf16* __restrict__ qkv,   // [Mtok, 3072]: q|k|v
    __bf16* __restrict__ attn_out) {  // [Mtok, 1024]
  extern __shared__ char smem[];
  float*  scorePart = reinterpret_cast<float*>(smem);              // [4][16][32]
  float*  mrow      = reinterpret_cast<float*>(smem + 8192);       // [16]
  float*  lrow      = mrow + 16;                                   // [16]
  float*  frow      = lrow + 16;                                   // [16]
  __bf16* Pt        = reinterpret_cast<__bf16*>(frow + 16);        // [16][40]
  __bf16* Vt        = reinterpret_cast<__bf16*>(smem + 8192 + 192 + 1280); // [4][256][32]

  const int bidx = blockIdx.x;               // 0 .. B*S/16-1
  const int b    = bidx >> 7;                // / (S/16)=128
  const int q0   = (bidx & 127) * 16;
  const int tid  = threadIdx.x;
  const int wave = tid >> 5;
  const int lane = tid & 31;
  const int lm   = lane & 15;
  const int lh   = lane >> 4;
  const int d0   = wave * DSLICE;
  const size_t rowQ = (size_t)b * Sc + q0;

  v8f zero8 = {0.f, 0.f, 0.f, 0.f, 0.f, 0.f, 0.f, 0.f};
  v8f acc[16];
#pragma unroll
  for (int j = 0; j < 16; ++j) acc[j] = zero8;

  if (tid < 16) { mrow[tid] = -3.0e38f; lrow[tid] = 0.0f; }
  __syncthreads();

  const int kb_start = q0 & ~31;   // first key block containing any t>=s
  for (int kb = kb_start; kb < Sc; kb += 32) {
    // --- stage V^T slice: lane handles token t=lane, scatters d->rows ---
    {
      const __bf16* vp = qkv + (size_t)((size_t)b * Sc + kb + lane) * NQKV
                       + 2 * Dc + d0;
#pragma unroll
      for (int c = 0; c < DSLICE; c += 8) {
        v8bf v = *reinterpret_cast<const v8bf*>(vp + c);
#pragma unroll
        for (int e = 0; e < 8; ++e)
          Vt[(wave * DSLICE + c + e) * 32 + lane] = v[e];
      }
    }
    // --- partial scores over this wave's 256-wide D slice ---
    v8f sc[2] = {zero8, zero8};
    for (int kk = 0; kk < DSLICE; kk += 32) {
      v16bf aq = load_frag(qkv + (rowQ + lm) * NQKV + d0 + kk, lh);
#pragma unroll
      for (int j = 0; j < 2; ++j) {
        v16bf bk = load_frag(
            qkv + (size_t)((size_t)b * Sc + kb + j * 16 + lm) * NQKV
                + Dc + d0 + kk, lh);
        sc[j] = wmma_bf16(aq, bk, sc[j]);
      }
    }
#pragma unroll
    for (int j = 0; j < 2; ++j)
#pragma unroll
      for (int r = 0; r < 8; ++r)
        scorePart[wave * 512 + (lh * 8 + r) * 32 + j * 16 + lm] = sc[j][r];
    __syncthreads();

    // --- online softmax (one thread per query row) ---
    if (tid < 16) {
      const int r = tid;
      const int sglob = q0 + r;
      float mold = mrow[r], lold = lrow[r];
      float mnew = mold;
      float sv[32];
#pragma unroll
      for (int t = 0; t < 32; ++t) {
        float s = scorePart[r * 32 + t] + scorePart[512 + r * 32 + t]
                + scorePart[1024 + r * 32 + t] + scorePart[1536 + r * 32 + t];
        s *= 0.5f;                       // 1/sqrt(B), B=4
        if (kb + t < sglob) s = 0.0f;    // triu: keep t >= s
        if (s == 0.0f) s = -9.0e15f;     // exact-zero mask
        sv[t] = s;
        mnew = fmaxf(mnew, s);
      }
      const float f = __expf(mold - mnew);
      float lsum = lold * f;
#pragma unroll
      for (int t = 0; t < 32; ++t) {
        float p = __expf(sv[t] - mnew);
        lsum += p;
        Pt[r * 40 + t] = (__bf16)p;
      }
      mrow[r] = mnew; lrow[r] = lsum; frow[r] = f;
    }
    __syncthreads();

    // --- rescale accumulators and P @ V over this key block ---
    float fr[8];
#pragma unroll
    for (int r = 0; r < 8; ++r) fr[r] = frow[lh * 8 + r];
    v16bf pf = load_frag(&Pt[lm * 40], lh);   // A-frag: P (16x32)
#pragma unroll
    for (int j = 0; j < 16; ++j) {
#pragma unroll
      for (int r = 0; r < 8; ++r) acc[j][r] *= fr[r];
      v16bf vf = load_frag(&Vt[(wave * DSLICE + j * 16 + lm) * 32], lh);
      acc[j] = wmma_bf16(pf, vf, acc[j]);
    }
    __syncthreads();   // protect LDS before next iteration restages
  }

  // --- normalize and emit bf16 attention output ---
  float lr[8];
#pragma unroll
  for (int r = 0; r < 8; ++r) lr[r] = 1.0f / lrow[lh * 8 + r];
#pragma unroll
  for (int j = 0; j < 16; ++j) {
    const int col = d0 + j * 16 + lm;
#pragma unroll
    for (int r = 0; r < 8; ++r) {
      const size_t row = rowQ + lh * 8 + r;
      attn_out[row * Dc + col] = (__bf16)(acc[j][r] * lr[r]);
    }
  }
}

// ---------------------------------------------------------------------------
extern "C" void kernel_launch(void* const* d_in, const int* in_sizes, int n_in,
                              void* d_out, int out_size, void* d_ws, size_t ws_size,
                              hipStream_t stream) {
  const float* x     = (const float*)d_in[0];
  const float* wqkv  = (const float*)d_in[1];
  const float* w_lin = (const float*)d_in[2];
  const float* b_lin = (const float*)d_in[3];
  const float* w_ff1 = (const float*)d_in[4];
  const float* b_ff1 = (const float*)d_in[5];
  const float* w_ff2 = (const float*)d_in[6];
  const float* b_ff2 = (const float*)d_in[7];
  float* out = (float*)d_out;

  char* ws = (char*)d_ws;
  size_t o = 0;
  auto take = [&](size_t bytes) {
    char* p = ws + o;
    o += (bytes + 255) & ~(size_t)255;
    return (void*)p;
  };
  __bf16* xb     = (__bf16*)take((size_t)Mtok * Dc * 2);
  __bf16* wqkvT  = (__bf16*)take((size_t)NQKV * Dc * 2);
  __bf16* wlinT  = (__bf16*)take((size_t)Dc * Dc * 2);
  __bf16* wff1T  = (__bf16*)take((size_t)Dc * Dc * 2);
  __bf16* wff2T  = (__bf16*)take((size_t)Dc * Dc * 2);
  __bf16* qkvb   = (__bf16*)take((size_t)Mtok * NQKV * 2);
  __bf16* attnb  = (__bf16*)take((size_t)Mtok * Dc * 2);
  float*  x2f    = (float*) take((size_t)Mtok * Dc * 4);
  __bf16* x2b    = (__bf16*)take((size_t)Mtok * Dc * 2);
  __bf16* hb     = (__bf16*)take((size_t)Mtok * Dc * 2);

  // 1) convert x to bf16
  {
    long n = (long)Mtok * Dc;
    cvt_bf16_kernel<<<dim3((unsigned)((n / 4 + 255) / 256)), dim3(256), 0, stream>>>(x, xb, n);
  }
  // 2) transpose+convert weights to [N,K] bf16
  transpose_cvt_kernel<<<dim3(NQKV / 32, Dc / 32), dim3(256), 0, stream>>>(wqkv,  wqkvT, Dc, NQKV);
  transpose_cvt_kernel<<<dim3(Dc / 32,  Dc / 32), dim3(256), 0, stream>>>(w_lin, wlinT, Dc, Dc);
  transpose_cvt_kernel<<<dim3(Dc / 32,  Dc / 32), dim3(256), 0, stream>>>(w_ff1, wff1T, Dc, Dc);
  transpose_cvt_kernel<<<dim3(Dc / 32,  Dc / 32), dim3(256), 0, stream>>>(w_ff2, wff2T, Dc, Dc);

  // 3) qkv = x @ wqkv  (bf16 out, row stride 3072)
  gemm_bf16_kernel<<<dim3(Mtok / TM, NQKV / TN), dim3(128), 0, stream>>>(
      xb, Dc, wqkvT, Dc, Dc, nullptr, nullptr, 0, nullptr, qkvb, NQKV);

  // 4) flash attention -> attnb (bf16)
  attn_flash_kernel<<<dim3(Mtok / 16), dim3(128), SMEM_ATTN, stream>>>(qkvb, attnb);

  // 5) x2 = attn_out @ w_lin + b_lin + x   (f32 + bf16 copies)
  gemm_bf16_kernel<<<dim3(Mtok / TM, Dc / TN), dim3(128), 0, stream>>>(
      attnb, Dc, wlinT, Dc, Dc, b_lin, x, 0, x2f, x2b, Dc);

  // 6) h = relu(x2 @ w_ff1 + b_ff1)  (bf16)
  gemm_bf16_kernel<<<dim3(Mtok / TM, Dc / TN), dim3(128), 0, stream>>>(
      x2b, Dc, wff1T, Dc, Dc, b_ff1, nullptr, 1, nullptr, hb, Dc);

  // 7) out = h @ w_ff2 + b_ff2 + x2  (f32)
  gemm_bf16_kernel<<<dim3(Mtok / TM, Dc / TN), dim3(128), 0, stream>>>(
      hb, Dc, wff2T, Dc, Dc, b_ff2, x2f, 0, out, nullptr, Dc);
}
